// GraphEncoder_36532991820399
// MI455X (gfx1250) — compile-verified, matrix operand order checked by reading, not run on hardware
//
#include <hip/hip_runtime.h>

// ---------------------------------------------------------------------------
// Types for gfx1250 WMMA (wave32)
// ---------------------------------------------------------------------------
typedef __attribute__((ext_vector_type(16))) __bf16 v16bf;
typedef __attribute__((ext_vector_type(8)))  __bf16 v8bf;
typedef __attribute__((ext_vector_type(4)))  __bf16 v4bf;
typedef __attribute__((ext_vector_type(8)))  float  v8f;
typedef __attribute__((ext_vector_type(4)))  float  v4f;

#define NNODES 4096
#define HDIM   128
#define NEDGE  8
#define PITCH  40   // bf16 halfwords per LDS row: 80B pitch, 16B aligned frags

__device__ __forceinline__ v4bf cvt4(v4f x) {
  v4bf r;
  r[0] = (__bf16)x[0]; r[1] = (__bf16)x[1];
  r[2] = (__bf16)x[2]; r[3] = (__bf16)x[3];
  return r;
}

// ---------------------------------------------------------------------------
// node_init = concat(emb_table[node_ids], node_states)   [N, 384]
// ---------------------------------------------------------------------------
__global__ void concat_init(const float* __restrict__ states,
                            const int*   __restrict__ ids,
                            const float* __restrict__ emb,
                            float* __restrict__ out) {
  const int n = blockIdx.x;
  const int t = threadIdx.x;           // 128 threads
  const int id = ids[n];
  out[(size_t)n * 384 + t]         = emb[(size_t)id * 128 + t];
  out[(size_t)n * 384 + 128 + t]   = states[(size_t)n * 256 + t];
  out[(size_t)n * 384 + 256 + t]   = states[(size_t)n * 256 + 128 + t];
}

// ---------------------------------------------------------------------------
// Generic tiled GEMM: C[z] = act(A[z] (xT?) @ B[z] + bias[z])
//   fp32 inputs -> bf16 tiles in LDS -> v_wmma_f32_16x16x32_bf16, f32 acc.
//   8 waves / 128x128 output tile; K stepped by 32.
//   Double-buffered LDS, one barrier per K-step; next tile's global_load_b128s
//   are issued before the WMMA chain so HBM latency overlaps the math.
// ---------------------------------------------------------------------------
__global__ __launch_bounds__(256)
void wmma_gemm(const float* __restrict__ A, long lda, long strideA, int transA,
               const float* __restrict__ B, long ldb, long strideB,
               const float* __restrict__ bias, long strideBias,
               float* __restrict__ C, long ldc, long strideC,
               int K, int doTanh) {
  __shared__ __align__(16) __bf16 ldsA[2][128][PITCH];  // [buf][row][k]
  __shared__ __align__(16) __bf16 ldsB[2][128][PITCH];  // [buf][col][k]

  const int tid  = threadIdx.x;
  const long rb  = blockIdx.x;
  const long cb  = blockIdx.y;
  const long z   = blockIdx.z;

  A += z * strideA;
  B += z * strideB;
  C += z * strideC;
  const float* bs = bias ? (bias + z * strideBias) : nullptr;

  const long rowBase = rb * 128;
  const long colBase = cb * 128;

  const int w     = tid >> 5;          // wave 0..7 -> 16-row strip
  const int lane  = tid & 31;
  const int mloc  = lane & 15;
  const int kbase = (lane >> 4) << 3;  // 0 or 8 (ISA 16-bit A/B frag layout)

  // per-thread staging coordinates
  const int anR = tid >> 3;            // no-trans A: 4 rows, stride 32
  const int anC = (tid & 7) * 4;       //             4 consecutive k
  const int atR = (tid & 31) * 4;      // trans A: 4 consecutive global cols
  const int atC = tid >> 5;            //          8 k per pass
  const int bR  = tid >> 5;            // B: 8 k rows per pass
  const int bC  = (tid & 31) * 4;      //    4 consecutive cols

  v4f ar[4], br[4];

  auto gload = [&](int k0) {
    if (!transA) {
      #pragma unroll
      for (int p = 0; p < 4; ++p)
        ar[p] = *(const v4f*)&A[(rowBase + anR + 32 * p) * lda + (k0 + anC)];
    } else {
      #pragma unroll
      for (int p = 0; p < 4; ++p)
        ar[p] = *(const v4f*)&A[(long)(k0 + atC + 8 * p) * lda + (rowBase + atR)];
    }
    #pragma unroll
    for (int p = 0; p < 4; ++p)
      br[p] = *(const v4f*)&B[(long)(k0 + bR + 8 * p) * ldb + (colBase + bC)];
  };

  auto lstore = [&](int buf) {
    if (!transA) {
      #pragma unroll
      for (int p = 0; p < 4; ++p)  // packed 8B stores along k
        *(v4bf*)&ldsA[buf][anR + 32 * p][anC] = cvt4(ar[p]);
    } else {
      #pragma unroll
      for (int p = 0; p < 4; ++p)
        #pragma unroll
        for (int q = 0; q < 4; ++q)
          ldsA[buf][atR + q][atC + 8 * p] = (__bf16)ar[p][q];
    }
    #pragma unroll
    for (int p = 0; p < 4; ++p)
      #pragma unroll
      for (int q = 0; q < 4; ++q)
        ldsB[buf][bC + q][bR + 8 * p] = (__bf16)br[p][q];
  };

  v8f acc[8] = {};
  const int nk = K >> 5;

  gload(0);                            // prologue: tile 0 -> regs
  for (int it = 0; it < nk; ++it) {
    const int buf = it & 1;
    lstore(buf);                       // regs (tile it) -> LDS[buf]
    __syncthreads();                   // stores visible; buf^1 reads retired
    if (it + 1 < nk) gload((it + 1) << 5);   // overlap HBM with WMMA below

    const int arow = w * 16 + mloc;
    v8bf alo = *(const v8bf*)&ldsA[buf][arow][kbase];
    v8bf ahi = *(const v8bf*)&ldsA[buf][arow][16 + kbase];
    v16bf afrag = __builtin_shufflevector(alo, ahi,
        0,1,2,3,4,5,6,7,8,9,10,11,12,13,14,15);

    v16bf bfrag[8];
    #pragma unroll
    for (int j = 0; j < 8; ++j) {      // 16 back-to-back ds_load_b128
      v8bf blo = *(const v8bf*)&ldsB[buf][j * 16 + mloc][kbase];
      v8bf bhi = *(const v8bf*)&ldsB[buf][j * 16 + mloc][16 + kbase];
      bfrag[j] = __builtin_shufflevector(blo, bhi,
          0,1,2,3,4,5,6,7,8,9,10,11,12,13,14,15);
    }
    #pragma unroll
    for (int j = 0; j < 8; ++j)
      acc[j] = __builtin_amdgcn_wmma_f32_16x16x32_bf16(
          false, afrag, false, bfrag[j], (short)0, acc[j], false, false);
  }

  // epilogue: C/D layout -> VGPR v: M = v + (lane<16?0:8), N = mloc
  const int rowOff = w * 16 + ((lane >> 4) << 3);
  #pragma unroll
  for (int j = 0; j < 8; ++j) {
    const int col = j * 16 + mloc;
    const float bv = bs ? bs[colBase + col] : 0.0f;
    #pragma unroll
    for (int v = 0; v < 8; ++v) {
      float val = acc[j][v] + bv;
      if (doTanh) val = tanhf(val);
      C[(rowBase + rowOff + v) * ldc + (colBase + col)] = val;
    }
  }
}

// ---------------------------------------------------------------------------
// Fused GRU gate math + 3x LayerNorm over H=128 (one row per 128-thread block)
// ---------------------------------------------------------------------------
__device__ __forceinline__ float blocksum128(float v, float* red) {
  #pragma unroll
  for (int o = 16; o; o >>= 1) v += __shfl_down(v, o, 32);
  __syncthreads();
  if ((threadIdx.x & 31) == 0) red[threadIdx.x >> 5] = v;
  __syncthreads();
  return red[0] + red[1] + red[2] + red[3];
}

__device__ __forceinline__ float ln128(float x, const float* g, const float* b,
                                       int c, float* red) {
  const float s  = blocksum128(x, red);
  const float s2 = blocksum128(x * x, red);
  const float mu  = s * (1.0f / 128.0f);
  const float var = s2 * (1.0f / 128.0f) - mu * mu;
  return (x - mu) * rsqrtf(var + 1e-5f) * g[c] + b[c];
}

__global__ __launch_bounds__(128)
void gru_ln(const float* __restrict__ gi, const float* __restrict__ gh,
            const float* __restrict__ h,
            const float* __restrict__ gamma, const float* __restrict__ beta,
            float* __restrict__ hout) {
  __shared__ float red[4];
  const int n = blockIdx.x;
  const int c = threadIdx.x;

  const float ir  = gi[(size_t)n * 384 + c];
  const float iz  = gi[(size_t)n * 384 + 128 + c];
  const float inn = gi[(size_t)n * 384 + 256 + c];
  const float hr  = gh[(size_t)n * 384 + c];
  const float hz  = gh[(size_t)n * 384 + 128 + c];
  const float hn  = gh[(size_t)n * 384 + 256 + c];
  const float hv  = h[(size_t)n * 128 + c];

  float r = ln128(ir + hr, gamma,       beta,       c, red);
  r = 1.0f / (1.0f + expf(-r));
  float zg = ln128(iz + hz, gamma + 128, beta + 128, c, red);
  zg = 1.0f / (1.0f + expf(-zg));
  float nn = ln128(inn + r * hn, gamma + 256, beta + 256, c, red);
  nn = tanhf(nn);

  hout[(size_t)n * 128 + c] = (1.0f - zg) * nn + zg * hv;
}

// ---------------------------------------------------------------------------
// Host-side orchestration (graph-capture safe: only kernel launches on stream)
// ---------------------------------------------------------------------------
extern "C" void kernel_launch(void* const* d_in, const int* in_sizes, int n_in,
                              void* d_out, int out_size, void* d_ws, size_t ws_size,
                              hipStream_t stream) {
  const float* adj        = (const float*)d_in[0];   // [E,N,N]
  const float* nodeStates = (const float*)d_in[1];   // [N,256]
  const int*   nodeIds    = (const int*)  d_in[2];   // [N]
  const float* embTable   = (const float*)d_in[3];   // [1000,128]
  const float* W_init     = (const float*)d_in[4];   // [384,128]
  const float* b_init     = (const float*)d_in[5];   // [128]
  const float* W_out      = (const float*)d_in[6];   // [E,128,128]
  const float* b_out      = (const float*)d_in[7];   // [E,128]
  const float* W_in       = (const float*)d_in[8];   // [E,128,128]
  const float* b_in       = (const float*)d_in[9];   // [E,128]
  const float* Wx         = (const float*)d_in[10];  // [2048,384]
  const float* bx         = (const float*)d_in[11];  // [384]
  const float* Wh         = (const float*)d_in[12];  // [128,384]
  const float* bh         = (const float*)d_in[13];  // [384]
  const float* lnG        = (const float*)d_in[14];  // [3,128]
  const float* lnB        = (const float*)d_in[15];  // [3,128]
  float* out = (float*)d_out;                        // [N,128]

  // workspace carve-out (fp32)
  char* ws = (char*)d_ws;
  float* nodeInit = (float*)ws; ws += (size_t)NNODES * 384 * 4;          // 6.3 MB
  float* h        = (float*)ws; ws += (size_t)NNODES * 128 * 4;          // 2.1 MB
  float* hoBuf    = (float*)ws; ws += (size_t)NEDGE * NNODES * 128 * 4;  // 16.8 MB
  float* hiBuf    = (float*)ws; ws += (size_t)NEDGE * NNODES * 128 * 4;  // 16.8 MB
  float* msg      = (float*)ws; ws += (size_t)NNODES * 2048 * 4;         // 33.6 MB
  float* gh       = (float*)ws; ws += (size_t)NNODES * 384 * 4;          // 6.3 MB
  float* gi       = nodeInit;   // nodeInit dead after h0 -> reuse for gi

  const dim3 blk(256);
  const long NN    = (long)NNODES * NNODES;   // adjacency batch stride
  const long NH    = (long)NNODES * 128;      // per-edge activation stride

  // h0 = tanh(node_init @ W_init + b_init)
  concat_init<<<NNODES, 128, 0, stream>>>(nodeStates, nodeIds, embTable, nodeInit);
  wmma_gemm<<<dim3(32, 1, 1), blk, 0, stream>>>(
      nodeInit, 384, 0, 0,  W_init, 128, 0,  b_init, 0,
      h, 128, 0,  384, 1);

  for (int t = 0; t < 2; ++t) {
    // ho[e] = tanh(h @ W_out[e] + b_out[e]) ; hi[e] likewise
    wmma_gemm<<<dim3(32, 1, NEDGE), blk, 0, stream>>>(
        h, 128, 0, 0,  W_out, 128, 128 * 128,  b_out, 128,
        hoBuf, 128, NH,  128, 1);
    wmma_gemm<<<dim3(32, 1, NEDGE), blk, 0, stream>>>(
        h, 128, 0, 0,  W_in, 128, 128 * 128,  b_in, 128,
        hiBuf, 128, NH,  128, 1);

    // mo[:, e*H:..] = A[e] @ ho[e]          (HBM-bound stream of A)
    wmma_gemm<<<dim3(32, 1, NEDGE), blk, 0, stream>>>(
        adj, NNODES, NN, 0,  hoBuf, 128, NH,  nullptr, 0,
        msg, 2048, 128,  NNODES, 0);
    // mi[:, e*H:..] = A[e]^T @ hi[e]
    wmma_gemm<<<dim3(32, 1, NEDGE), blk, 0, stream>>>(
        adj, NNODES, NN, 1,  hiBuf, 128, NH,  nullptr, 0,
        msg + 1024, 2048, 128,  NNODES, 0);

    // gi = msg @ Wx + bx ; gh = h @ Wh + bh
    wmma_gemm<<<dim3(32, 3, 1), blk, 0, stream>>>(
        msg, 2048, 0, 0,  Wx, 384, 0,  bx, 0,
        gi, 384, 0,  2048, 0);
    wmma_gemm<<<dim3(32, 3, 1), blk, 0, stream>>>(
        h, 128, 0, 0,  Wh, 384, 0,  bh, 0,
        gh, 384, 0,  128, 0);

    gru_ln<<<NNODES, 128, 0, stream>>>(gi, gh, h, lnG, lnB,
                                       (t == 1) ? out : h);
  }
  (void)in_sizes; (void)n_in; (void)out_size; (void)ws_size;
}
